// DynamicConv2D_39719857554079
// MI455X (gfx1250) — matrix operand order repaired and checked by reading, not compiled, not run on hardware
//
#include <hip/hip_runtime.h>
#include <math.h>

// ---------------- problem constants ----------------
#define BB   32
#define HH   128
#define WW   128
#define CIN  64
#define FF   64
#define NK   4
#define HID  16
#define TEMP 30.0f

#define KSTEPS 18           // 576 / 32
#define WFRAG_UINTS 18432   // per-sample packed f16 weight fragments (36864 halves)
#define PIXF 72             // tile pixel stride in elements (64 + 8 pad)
#define XH_OFF (4u << 20)   // byte offset of f16 image copy in workspace

typedef __attribute__((ext_vector_type(16))) _Float16 v16h;
typedef __attribute__((ext_vector_type(8)))  _Float16 v8h;
typedef __attribute__((ext_vector_type(8)))  float    v8f;
typedef __attribute__((ext_vector_type(16))) float    v16f;
typedef __attribute__((ext_vector_type(4)))  unsigned v4u;
typedef __attribute__((ext_vector_type(8)))  int      v8i;
typedef __attribute__((ext_vector_type(4)))  int      v4i;

#if defined(__gfx1250__) && __has_builtin(__builtin_amdgcn_tensor_load_to_lds)
#define HAVE_TDM 1
#else
#define HAVE_TDM 0
#endif

// ---------------- pass 1a: pooling only (fp32 fallback path) ----------------
__global__ __launch_bounds__(256) void pool_kernel(const float* __restrict__ x,
                                                   float* __restrict__ pooled) {
    int blk = blockIdx.x;            // b*128 + h
    int b = blk >> 7, h = blk & 127;
    int t = threadIdx.x;
    int c = t & 63, wq = t >> 6;
    const float* p = x + (((b * HH + h) * WW) << 6);
    float s = 0.f;
    int w0 = wq << 5;
    #pragma unroll 4
    for (int w = w0; w < w0 + 32; ++w) s += p[(w << 6) + c];
    __shared__ float red[256];
    red[t] = s;
    __syncthreads();
    if (wq == 0) {
        float tot = red[c] + red[64 + c] + red[128 + c] + red[192 + c];
        atomicAdd(&pooled[(b << 6) + c], tot);
    }
}

// ---------------- pass 1b: fused pooling + f32->f16 image convert ----------------
__global__ __launch_bounds__(256) void pool_cvt_kernel(const float* __restrict__ x,
                                                       float* __restrict__ pooled,
                                                       unsigned* __restrict__ xh) {
    int blk = blockIdx.x;            // b*128 + h
    int b = blk >> 7, h = blk & 127;
    int t = threadIdx.x;
    const float2* row = (const float2*)(x + (((b * HH + h) * WW) << 6)); // 4096 float2
    unsigned* orow = xh + (((b * HH + h) * WW) << 5);                    // 4096 packed-f16 pairs
    float s0 = 0.f, s1 = 0.f;
    #pragma unroll
    for (int i = 0; i < 16; ++i) {
        int j = t + (i << 8);
        float2 v = row[j];
        s0 += v.x; s1 += v.y;
        union { unsigned u; _Float16 hh[2]; } pk;
        pk.hh[0] = (_Float16)v.x;
        pk.hh[1] = (_Float16)v.y;
        orow[j] = pk.u;
    }
    // j & 31 == t & 31 for all iterations -> thread owns channel pair cp = t&31
    __shared__ float r0[256], r1[256];
    r0[(t & 31) * 8 + (t >> 5)] = s0;
    r1[(t & 31) * 8 + (t >> 5)] = s1;
    __syncthreads();
    if (t < 64) {
        int cp = t >> 1, which = t & 1;
        const float* r = which ? r1 : r0;
        float s = 0.f;
        #pragma unroll
        for (int q = 0; q < 8; ++q) s += r[cp * 8 + q];
        atomicAdd(&pooled[(b << 6) + (cp << 1) + which], s);
    }
}

// ---------------- pass 2: attention MLP + softmax + bias agg ----------------
__global__ __launch_bounds__(32) void attn_kernel(const float* __restrict__ pooled,
                                                  const float* __restrict__ w1,
                                                  const float* __restrict__ b1,
                                                  const float* __restrict__ w2,
                                                  const float* __restrict__ b2,
                                                  const float* __restrict__ bk,
                                                  float* __restrict__ pi_out,
                                                  float* __restrict__ bagg_out) {
    int b = threadIdx.x;
    if (b >= BB) return;
    float pm[CIN];
    const float inv = 1.0f / (float)(HH * WW);
    for (int c = 0; c < CIN; ++c) pm[c] = pooled[(b << 6) + c] * inv;
    float hb[HID];
    for (int j = 0; j < HID; ++j) {
        float s = b1[j];
        for (int c = 0; c < CIN; ++c) s += pm[c] * w1[c * HID + j];
        hb[j] = s > 0.f ? s : 0.f;
    }
    float lg[NK];
    float mx = -1e30f;
    for (int k = 0; k < NK; ++k) {
        float s = b2[k];
        for (int j = 0; j < HID; ++j) s += hb[j] * w2[j * NK + k];
        lg[k] = s / TEMP;
        mx = fmaxf(mx, lg[k]);
    }
    float den = 0.f;
    for (int k = 0; k < NK; ++k) { lg[k] = expf(lg[k] - mx); den += lg[k]; }
    float pi[NK];
    for (int k = 0; k < NK; ++k) { pi[k] = lg[k] / den; pi_out[b * NK + k] = pi[k]; }
    for (int f = 0; f < FF; ++f) {
        float s = 0.f;
        for (int k = 0; k < NK; ++k) s += pi[k] * bk[k * FF + f];
        bagg_out[(b << 6) + f] = s;
    }
}

// ---------------- pass 3: aggregate weights into WMMA B-fragment order ----------------
__global__ __launch_bounds__(256) void wagg_kernel(const float* __restrict__ Wk,
                                                   const float* __restrict__ pi,
                                                   unsigned* __restrict__ wfrag) {
    int b = blockIdx.x;
    float p0 = pi[b * NK + 0], p1 = pi[b * NK + 1];
    float p2 = pi[b * NK + 2], p3 = pi[b * NK + 3];
    unsigned* out = wfrag + b * WFRAG_UINTS;
    const int KS = 9 * CIN * FF;  // 36864 floats per base kernel
    for (int idx = threadIdx.x; idx < WFRAG_UINTS; idx += 256) {
        int u2   = idx & 3;
        int lane = (idx >> 2) & 31;
        int hh   = (idx >> 7) & 1;
        int nt   = (idx >> 8) & 3;
        int ks   = idx >> 10;
        int u    = (hh << 2) + u2;
        int n    = (nt << 4) + (lane & 15);
        int kk   = (ks << 5) + ((lane >> 4) << 3) + (hh << 4) + (u2 << 1);
        int c    = kk & 63;
        int p    = kk >> 6;
        int base = (p << 12) + (c << 6) + n;  // p*64*64 + c*64 + n
        float v0 = p0 * Wk[base]               + p1 * Wk[base + KS]
                 + p2 * Wk[base + 2 * KS]      + p3 * Wk[base + 3 * KS];
        float v1 = p0 * Wk[base + 64]          + p1 * Wk[base + KS + 64]
                 + p2 * Wk[base + 2 * KS + 64] + p3 * Wk[base + 3 * KS + 64];
        union { unsigned u32; _Float16 h[2]; } pk;
        pk.h[0] = (_Float16)v0;
        pk.h[1] = (_Float16)v1;
        out[idx] = pk.u32;
    }
}

// ---------------- pass 4: implicit-GEMM conv via WMMA + TDM tile staging ----------------
// F16T=true : xin is a pre-converted f16 image; TDM copies raw f16 tiles; no cvt in hot loop.
// F16T=false: xin is the fp32 image; TDM copies fp32 tiles; A converted in-register.
template<bool F16T>
__global__ __launch_bounds__(256) void conv_kernel(const void* __restrict__ xin,
                                                   const unsigned* __restrict__ wfrag,
                                                   const float* __restrict__ bagg,
                                                   float* __restrict__ y) {
    __shared__ unsigned wlds[WFRAG_UINTS];                            // 73728 B B-fragments
    __shared__ __align__(64) unsigned char tileraw[F16T ? 180 * 144 : 180 * 288];

    const int b      = blockIdx.z;
    const int h0     = blockIdx.y << 3;
    const int wstrip = blockIdx.x << 6;
    const int t      = threadIdx.x;
    const int lane   = t & 31;
    const int wave   = t >> 5;

    // load packed weights for this sample into LDS (128-bit, coalesced)
    {
        const uint4* src = (const uint4*)(wfrag + b * WFRAG_UINTS);
        uint4* dst = (uint4*)wlds;
        #pragma unroll
        for (int i = 0; i < 18; ++i) dst[t + (i << 8)] = src[t + (i << 8)];
    }

    float bias[4];
    #pragma unroll
    for (int nt = 0; nt < 4; ++nt) bias[nt] = bagg[(b << 6) + (nt << 4) + (lane & 15)];

    const int m  = lane & 15;            // A row within wave's 16 positions
    const int ab = (lane >> 4) << 3;     // A K-sub-base (0 or 8 elements)
    const _Float16* wh = (const _Float16*)wlds;
    const bool hInterior = (blockIdx.y >= 1) && (blockIdx.y <= 14);

    static const int DH[9] = {0,0,0,1,1,1,2,2,2};
    static const int DW[9] = {0,1,2,0,1,2,0,1,2};

    for (int wsub = 0; wsub < 4; ++wsub) {
        const int wbase = wstrip + (wsub << 4);
        __syncthreads();  // previous compute finished; weights visible on first iter

#if HAVE_TDM
        const bool interior = hInterior && (wbase >= 1) && (wbase <= 111);
#else
        const bool interior = false;
#endif
        if (!interior) {
            // manual zero-padded staging (image borders)
            if (F16T) {
                const unsigned* xh32 = (const unsigned*)xin;   // packed f16 pairs
                unsigned* tile32 = (unsigned*)tileraw;
                for (int idx = t; idx < 180 * 32; idx += 256) {
                    int c2  = idx & 31;
                    int pix = idx >> 5;
                    int ih  = pix / 18;
                    int iw  = pix - ih * 18;
                    int gh  = h0 - 1 + ih;
                    int gw  = wbase - 1 + iw;
                    unsigned v = 0u;
                    if ((unsigned)gh < (unsigned)HH && (unsigned)gw < (unsigned)WW)
                        v = xh32[((((b * HH + gh) * WW) + gw) << 5) + c2];
                    tile32[pix * (PIXF / 2) + c2] = v;
                }
            } else {
                const float* xf = (const float*)xin;
                float* tilef = (float*)tileraw;
                for (int idx = t; idx < 180 * 32; idx += 256) {
                    int c2  = idx & 31;
                    int pix = idx >> 5;
                    int ih  = pix / 18;
                    int iw  = pix - ih * 18;
                    int gh  = h0 - 1 + ih;
                    int gw  = wbase - 1 + iw;
                    float2 v = make_float2(0.f, 0.f);
                    if ((unsigned)gh < (unsigned)HH && (unsigned)gw < (unsigned)WW)
                        v = *(const float2*)(xf + ((((b * HH + gh) * WW) + gw) << 6) + (c2 << 1));
                    *(float2*)&tilef[pix * PIXF + (c2 << 1)] = v;
                }
            }
        }
#if HAVE_TDM
        else if (wave == 0) {
            // 2-D TDM: one halo row = 18*64 contiguous elements, 10 rows, stride 128*64.
            // LDS pad keeps the 72-element pixel stride: f16: +4DW/32DW; fp32: +8DW/64DW.
            unsigned long long ga;
            int d1w0;
            if (F16T) {
                ga = (unsigned long long)((const _Float16*)xin +
                     ((((b * HH + (h0 - 1)) * WW) + (wbase - 1)) << 6));
                d1w0 = (1 << 16) | (1 << 20) | (4 << 22) | (3 << 25);
            } else {
                ga = (unsigned long long)((const float*)xin +
                     ((((b * HH + (h0 - 1)) * WW) + (wbase - 1)) << 6));
                d1w0 = (2 << 16) | (1 << 20) | (5 << 22) | (7 << 25);
            }
            v4u g0;
            g0.x = 1u;                                        // count=1, user mode
            g0.y = (unsigned)(unsigned long long)(&tileraw[0]); // lds_addr
            g0.z = (unsigned)(ga & 0xFFFFFFFFull);
            g0.w = (unsigned)((ga >> 32) & 0x1FFFFFFull) | (2u << 30);  // [56:32] | type=2
            v8i g1;
            g1[0] = d1w0;
            g1[1] = 0;                                        // abar=0, tdim0.lo16=0
            g1[2] = 0x4000;                                   // tdim0=1<<30, tdim1.lo16=0
            g1[3] = 0x4000 | (1152 << 16);                    // tdim1 hi16, tile_dim0=1152
            g1[4] = 10;                                       // tile_dim1=10, tile_dim2=0
            g1[5] = WW * CIN;                                 // tensor_dim0_stride = 8192
            g1[6] = 0;
            g1[7] = 0;
            v4i g2 = {0, 0, 0, 0};
            v4i g3 = {0, 0, 0, 0};
            v8i g4 = {0, 0, 0, 0, 0, 0, 0, 0};
            __builtin_amdgcn_tensor_load_to_lds(g0, g1, g2, g3, g4, 0);
            __builtin_amdgcn_s_wait_tensorcnt(0);
        }
#endif
        __syncthreads();

        v8f acc[4];
        #pragma unroll
        for (int nt = 0; nt < 4; ++nt) acc[nt] = (v8f){0.f,0.f,0.f,0.f,0.f,0.f,0.f,0.f};

        union AU { v16f v; v8f h[2]; };
        AU Af[2];           // double-buffered to break WMMA->VALU WAR hazards (fp32 path)
        v16h Av[2];

        #pragma unroll
        for (int ks = 0; ks < KSTEPS; ++ks) {
            const int p    = ks >> 1;
            const int c0   = (ks & 1) << 5;
            const int pixb = (wave + DH[p]) * 18 + m + DW[p];
            const int aoff = pixb * PIXF + c0 + ab;
            const int sel  = ks & 1;
            if (F16T) {
                const _Float16* th = (const _Float16*)tileraw;
                union { v16h v; v8h h[2]; } A;
                A.h[0] = *(const v8h*)(th + aoff);
                A.h[1] = *(const v8h*)(th + aoff + 16);
                Av[sel] = A.v;
            } else {
                const float* tf = (const float*)tileraw;
                Af[sel].h[0] = *(const v8f*)(tf + aoff);
                Af[sel].h[1] = *(const v8f*)(tf + aoff + 16);
                Av[sel] = __builtin_convertvector(Af[sel].v, v16h);
            }
            #pragma unroll
            for (int nt = 0; nt < 4; ++nt) {
                const int boff = ((((ks << 2) + nt) << 1) * 32 + lane) << 3;
                union { v16h v; v8h h[2]; } Bf;
                Bf.h[0] = *(const v8h*)(wh + boff);
                Bf.h[1] = *(const v8h*)(wh + boff + 256);
                acc[nt] = __builtin_amdgcn_wmma_f32_16x16x32_f16(
                    false, Av[sel], false, Bf.v, (short)0, acc[nt], false, false);
            }
        }

        // write 8x16x64 outputs (+bias). C layout: VGPR v -> M = v + (lane>>4)*8, N = lane&15.
        const int hrow = h0 + wave;
        #pragma unroll
        for (int nt = 0; nt < 4; ++nt) {
            const int f = (nt << 4) + (lane & 15);
            #pragma unroll
            for (int v = 0; v < 8; ++v) {
                const int M    = v + ((lane >> 4) << 3);
                const int wcol = wbase + M;
                y[(((b * HH + hrow) * WW + wcol) << 6) + f] = acc[nt][v] + bias[nt];
            }
        }
    }
}

// ---------------- launcher ----------------
extern "C" void kernel_launch(void* const* d_in, const int* in_sizes, int n_in,
                              void* d_out, int out_size, void* d_ws, size_t ws_size,
                              hipStream_t stream) {
    const float* x   = (const float*)d_in[0];
    const float* Wk  = (const float*)d_in[1];
    const float* bk  = (const float*)d_in[2];
    const float* w1  = (const float*)d_in[3];
    const float* b1  = (const float*)d_in[4];
    const float* w2  = (const float*)d_in[5];
    const float* b2  = (const float*)d_in[6];
    float* y = (float*)d_out;

    float* pooled = (float*)d_ws;                          // [32][64]
    float* pi     = pooled + BB * CIN;                     // [32][4]
    float* bagg   = pi + BB * NK;                          // [32][64]
    unsigned* wfrag = (unsigned*)((char*)d_ws + 32768);    // [32][18432] packed f16 pairs

    const size_t xh_bytes = (size_t)BB * HH * WW * CIN * 2;
    const bool f16path = ws_size >= (size_t)XH_OFF + xh_bytes;

    (void)hipMemsetAsync(d_ws, 0, BB * CIN * sizeof(float), stream);

    if (f16path) {
        unsigned* xh = (unsigned*)((char*)d_ws + XH_OFF);
        pool_cvt_kernel<<<BB * HH, 256, 0, stream>>>(x, pooled, xh);
        attn_kernel<<<1, 32, 0, stream>>>(pooled, w1, b1, w2, b2, bk, pi, bagg);
        wagg_kernel<<<BB, 256, 0, stream>>>(Wk, pi, wfrag);
        conv_kernel<true><<<dim3(2, HH / 8, BB), 256, 0, stream>>>((const void*)xh, wfrag, bagg, y);
    } else {
        pool_kernel<<<BB * HH, 256, 0, stream>>>(x, pooled);
        attn_kernel<<<1, 32, 0, stream>>>(pooled, w1, b1, w2, b2, bk, pi, bagg);
        wagg_kernel<<<BB, 256, 0, stream>>>(Wk, pi, wfrag);
        conv_kernel<false><<<dim3(2, HH / 8, BB), 256, 0, stream>>>((const void*)x, wfrag, bagg, y);
    }
}
